// LM_1657857376766
// MI455X (gfx1250) — matrix operand order repaired and checked by reading, not compiled, not run on hardware
//
#include <hip/hip_runtime.h>
#include <hip/hip_bf16.h>
#include <math.h>

typedef __attribute__((ext_vector_type(16))) __bf16 v16bf;
typedef __attribute__((ext_vector_type(8)))  float  v8f;
typedef __attribute__((ext_vector_type(4)))  unsigned uint4v;
typedef __attribute__((ext_vector_type(8)))  int    int8v;
typedef __attribute__((ext_vector_type(4)))  int    int4v;

#define B_   4
#define T_   128
#define V_   32000
#define H_   512
#define D_   4
#define MDEC (B_*(T_-D_))   // 496

__device__ __forceinline__ unsigned short f2bf(float f) {
  union { float f; unsigned u; } v; v.f = f;
  unsigned u = v.u;
  u += 0x7FFFu + ((u >> 16) & 1u);   // round-to-nearest-even
  return (unsigned short)(u >> 16);
}

// ---------------------------------------------------------------------------
// TDM 2D tile load: copies a tile_dim1 x tile_dim0 (rows x contiguous elems)
// bf16 tile from global into LDS. Padding config makes the TDM emit rows with
// a 36-ushort (72B) LDS stride: 32 elems (64B = 16 DWORDs, pad_interval=3)
// followed by 2 DWORDs of pad (pad_amount=1). OOB rows/cols read as zero.
// Descriptor layout per CDNA5 ISA section 8 (D# groups 0/1; later groups
// unused for 2D tiles -> zeros). 6-arg builtin form (clang-23 toolchain).
// ---------------------------------------------------------------------------
__device__ __forceinline__ void tdm_load_2d(unsigned lds_off,
                                            const unsigned short* gptr,
                                            unsigned dim0_rem,   // elems left in K
                                            unsigned dim1_rem,   // rows left
                                            unsigned stride0) {  // row stride, elems
  unsigned long long ga = (unsigned long long)(size_t)gptr;
  union { uint4v v; unsigned u[4]; } g0;
  union { int8v  v; unsigned u[8]; } g1;
  g0.u[0] = 1u;                                   // count=1, user mode
  g0.u[1] = lds_off;                              // lds_addr (bytes)
  g0.u[2] = (unsigned)(ga & 0xFFFFFFFFu);         // global_addr[31:0]
  g0.u[3] = (unsigned)((ga >> 32) & 0x01FFFFFFu)  // global_addr[56:32]
          | (2u << 30);                           // type=2 ("image")
  g1.u[0] = (1u << 16)                            // data_size = 1 -> 2 bytes
          | (1u << 20)                            // pad_enable
          | (3u << 22)                            // pad_interval: 16 DWORDs
          | (1u << 25);                           // pad_amount:   2 DWORDs
  g1.u[1] = (dim0_rem & 0xFFFFu) << 16;           // tensor_dim0[15:0]
  g1.u[2] = (dim0_rem >> 16) | ((dim1_rem & 0xFFFFu) << 16); // dim0 hi | dim1 lo
  g1.u[3] = (dim1_rem >> 16) | (32u << 16);       // dim1 hi | tile_dim0 = 32
  g1.u[4] = 64u;                                  // tile_dim1 = 64, tile_dim2 = 0
  g1.u[5] = stride0;                              // tensor_dim0_stride[31:0]
  g1.u[6] = 0u;                                   // stride hi / dim1_stride lo
  g1.u[7] = 0u;
  int4v z4 = {};
  int8v z8 = {};
  __builtin_amdgcn_tensor_load_to_lds(g0.v, g1.v, z4, z4, z8, 0);
}

// ---------------------------------------------------------------------------
// GEMM: C[m,n] = sum_k A[m,k]*B[n,k] (+ bias[n])   -- all-NT, bf16 operands
//   A: bf16 [M,K] (lda, batch stride sA);  B: bf16 [N,K] (ldb, batch stride sB)
//   C: f32 (nullable) and/or Cbf: bf16 (nullable), both ldc / sC
// Tiles staged by the Tensor Data Mover (wave 0 issues, s_wait_tensorcnt).
// Block tile 64x64, K-step 32, 8 waves, each wave owns 16x32 of C via two
// v_wmma_f32_16x16x32_bf16 accumulators.
// ---------------------------------------------------------------------------
#define BM 64
#define BN 64
#define BK 32
#define LSTR 36   // LDS row stride (ushorts) produced by the TDM pad config

__global__ __launch_bounds__(256) void gemm_tdm_bf16(
    const unsigned short* __restrict__ A, size_t sA, int lda,
    const unsigned short* __restrict__ B, size_t sB, int ldb,
    float* __restrict__ C, size_t sC, int ldc,
    unsigned short* __restrict__ Cbf,
    const float* __restrict__ bias, int M, int N, int K)
{
  __shared__ unsigned short As[BM * LSTR];
  __shared__ unsigned short Bs[BN * LSTR];

  const int tid   = threadIdx.x;
  const int lane  = tid & 31;
  const int wave  = tid >> 5;
  const int waveM = wave & 3;
  const int waveN = wave >> 2;
  const int hi    = lane >> 4;
  const int ml    = lane & 15;

  const int blockN = blockIdx.x * BN;
  const int blockM = blockIdx.y * BM;
  const int batch  = blockIdx.z;

  const unsigned short* Ab = A + (size_t)batch * sA + (size_t)blockM * lda;
  const unsigned short* Bb = B + (size_t)batch * sB + (size_t)blockN * ldb;

  // LDS byte offsets: generic-address low 32 bits == LDS offset (ISA 10.2).
  const unsigned As_off = (unsigned)(size_t)As;
  const unsigned Bs_off = (unsigned)(size_t)Bs;

  v8f acc[2] = {};

  for (int kt = 0; kt < K; kt += BK) {
    __syncthreads();                       // prior frag reads done
    if (wave == 0) {
      tdm_load_2d(As_off, Ab + kt, (unsigned)(K - kt), (unsigned)(M - blockM),
                  (unsigned)lda);
      tdm_load_2d(Bs_off, Bb + kt, (unsigned)(K - kt), (unsigned)(N - blockN),
                  (unsigned)ldb);
      __builtin_amdgcn_s_wait_tensorcnt(0);
    }
    __syncthreads();                       // tiles visible to all waves

    union { v16bf v; unsigned u[8]; } af, bf0, bf1;
    const unsigned short* Arow = &As[(waveM * 16 + ml) * LSTR];
    #pragma unroll
    for (int j = 0; j < 8; ++j) {
      // A 16x32 bf16: VGPR j -> K = 16*hi + 2j (contiguous, matches B)
      int koff = 16 * hi + 2 * j;
      af.u[j] = *(const unsigned*)(Arow + koff);
    }
    const unsigned short* Brow0 = &Bs[(waveN * 32 +      ml) * LSTR + 16 * hi];
    const unsigned short* Brow1 = &Bs[(waveN * 32 + 16 + ml) * LSTR + 16 * hi];
    #pragma unroll
    for (int j = 0; j < 8; ++j) {
      bf0.u[j] = *(const unsigned*)(Brow0 + 2 * j);
      bf1.u[j] = *(const unsigned*)(Brow1 + 2 * j);
    }
    acc[0] = __builtin_amdgcn_wmma_f32_16x16x32_bf16(false, af.v, false, bf0.v,
                                                     (short)0, acc[0], false, false);
    acc[1] = __builtin_amdgcn_wmma_f32_16x16x32_bf16(false, af.v, false, bf1.v,
                                                     (short)0, acc[1], false, false);
  }

  float*          Cb  = C   ? C   + (size_t)batch * sC : nullptr;
  unsigned short* Cbb = Cbf ? Cbf + (size_t)batch * sC : nullptr;
  #pragma unroll
  for (int t = 0; t < 2; ++t) {
    int col = blockN + waveN * 32 + t * 16 + ml;   // N is a multiple of 64 here
    float bv = bias ? bias[col] : 0.0f;
    #pragma unroll
    for (int j = 0; j < 8; ++j) {
      int row = blockM + waveM * 16 + 8 * hi + j;  // C layout: M = vgpr + 8*hi
      if (row < M) {
        float val = acc[t][j] + bv;
        if (Cb)  Cb [(size_t)row * ldc + col] = val;
        if (Cbb) Cbb[(size_t)row * ldc + col] = f2bf(val);
      }
    }
  }
}

// ---------------------------------------------------------------------------
// Helper kernels
// ---------------------------------------------------------------------------
__global__ void cvt_bf16(const float* __restrict__ in,
                         unsigned short* __restrict__ out, int n) {
  int i = blockIdx.x * blockDim.x + threadIdx.x;
  if (i < n) out[i] = f2bf(in[i]);
}

__global__ void embed_gather(const float* __restrict__ emb,
                             const int* __restrict__ x,
                             float* __restrict__ outf,
                             unsigned short* __restrict__ outb) {
  int bt = blockIdx.x;
  int tok = x[bt];
  const float* src = emb + (size_t)tok * H_;
  for (int c = threadIdx.x; c < H_; c += blockDim.x) {
    float v = src[c];
    outf[(size_t)bt * H_ + c] = v;
    outb[(size_t)bt * H_ + c] = f2bf(v);
  }
}

__global__ void transpose_g(const float* __restrict__ G,
                            unsigned short* __restrict__ Gt, int l) {
  int i = blockIdx.x * blockDim.x + threadIdx.x;   // B*T*T
  if (i >= B_ * T_ * T_) return;
  int b = i >> 14, r = (i >> 7) & 127, c = i & 127;
  Gt[i] = f2bf(G[(((size_t)b * 2 + l) * T_ + c) * T_ + r]);
}

// curT[b][h][j] = cur[b][j][h]  (f32 in -> bf16 out), makes the einsum NT.
__global__ void transpose_act(const float* __restrict__ in,
                              unsigned short* __restrict__ outT) {
  int i = blockIdx.x * blockDim.x + threadIdx.x;   // B*H*T = 262144
  if (i >= B_ * H_ * T_) return;
  int b = i >> 16, h = (i >> 7) & (H_ - 1), j = i & (T_ - 1);
  outT[i] = f2bf(in[(((size_t)b * T_ + j) << 9) + h]);
}

__global__ void gru_gate(const float* __restrict__ gi, const float* __restrict__ gh,
                         const float* __restrict__ bih, const float* __restrict__ bhh,
                         const float* __restrict__ hprev,
                         float* __restrict__ hout, unsigned short* __restrict__ houtb,
                         int Mrows) {
  int i = blockIdx.x * blockDim.x + threadIdx.x;
  if (i >= Mrows * H_) return;
  int m = i >> 9, c = i & (H_ - 1);
  size_t g = (size_t)m * (3 * H_);
  float ir = gi[g + c]          + bih[c];
  float iz = gi[g + H_ + c]     + bih[H_ + c];
  float in = gi[g + 2 * H_ + c] + bih[2 * H_ + c];
  float hr = gh[g + c]          + bhh[c];
  float hz = gh[g + H_ + c]     + bhh[H_ + c];
  float hn = gh[g + 2 * H_ + c] + bhh[2 * H_ + c];
  float r = 1.0f / (1.0f + expf(-(ir + hr)));
  float z = 1.0f / (1.0f + expf(-(iz + hz)));
  float n = tanhf(in + r * hn);
  float h = (1.0f - z) * n + z * hprev[i];
  hout[i]  = h;
  houtb[i] = f2bf(h);
}

__global__ void copy_h0(const float* __restrict__ inf,
                        float* __restrict__ h0f, unsigned short* __restrict__ h0b) {
  int i = blockIdx.x * blockDim.x + threadIdx.x;
  if (i >= MDEC * H_) return;
  int m = i >> 9, c = i & (H_ - 1);
  int b = m / (T_ - D_), t = m % (T_ - D_);
  float v = inf[(((size_t)b * T_ + t) << 9) + c];
  h0f[i] = v;
  h0b[i] = f2bf(v);
}

__global__ void init_next_in(const float* __restrict__ embedded,
                             unsigned short* __restrict__ ninb) {
  int i = blockIdx.x * blockDim.x + threadIdx.x;
  if (i >= MDEC * H_) return;
  int m = i >> 9, c = i & (H_ - 1);
  int b = m / (T_ - D_), t = m % (T_ - D_);
  int st = (t == 0) ? (T_ - 1) : (t - 1);
  ninb[i] = f2bf(embedded[(((size_t)b * T_ + st) << 9) + c]);
}

__global__ void argmax_rows(const float* __restrict__ logits, int ldr,
                            int* __restrict__ idx) {
  __shared__ float sv[256];
  __shared__ int   si[256];
  int m = blockIdx.x;
  const float* row = logits + (size_t)m * ldr;
  float best = -3.4e38f; int bi = 0;
  for (int v = threadIdx.x; v < V_; v += 256) {
    float f = row[v];
    if (f > best) { best = f; bi = v; }   // ascending scan keeps first max
  }
  sv[threadIdx.x] = best; si[threadIdx.x] = bi;
  __syncthreads();
  for (int s = 128; s > 0; s >>= 1) {
    if (threadIdx.x < s) {
      float ov = sv[threadIdx.x + s]; int oi = si[threadIdx.x + s];
      if (ov > sv[threadIdx.x] ||
          (ov == sv[threadIdx.x] && oi < si[threadIdx.x])) {
        sv[threadIdx.x] = ov; si[threadIdx.x] = oi;
      }
    }
    __syncthreads();
  }
  if (threadIdx.x == 0) idx[m] = si[0];
}

__global__ void gather_next(const float* __restrict__ emb,
                            const int* __restrict__ idx,
                            unsigned short* __restrict__ ninb) {
  int m = blockIdx.x;
  int tok = idx[m];
  const float* src = emb + (size_t)tok * H_;
  for (int c = threadIdx.x; c < H_; c += blockDim.x)
    ninb[(size_t)m * H_ + c] = f2bf(src[c]);
}

// ---------------------------------------------------------------------------
// Host orchestration
// ---------------------------------------------------------------------------
extern "C" void kernel_launch(void* const* d_in, const int* in_sizes, int n_in,
                              void* d_out, int out_size, void* d_ws, size_t ws_size,
                              hipStream_t stream) {
  (void)in_sizes; (void)n_in; (void)out_size; (void)ws_size;
  const int*   x    = (const int*)  d_in[0];
  const float* G    = (const float*)d_in[1];
  const float* emb  = (const float*)d_in[2];
  const float* eWih = (const float*)d_in[3];
  const float* eWhh = (const float*)d_in[4];
  const float* ebih = (const float*)d_in[5];
  const float* ebhh = (const float*)d_in[6];
  const float* dWih = (const float*)d_in[7];
  const float* dWhh = (const float*)d_in[8];
  const float* dbih = (const float*)d_in[9];
  const float* dbhh = (const float*)d_in[10];
  const float* ffW  = (const float*)d_in[11];
  const float* ffb  = (const float*)d_in[12];
  float* out = (float*)d_out;

  char* w = (char*)d_ws;
  size_t off = 0;
  auto allocB = [&](size_t nbytes) -> void* {
    void* p = (void*)(w + off);
    off += ((nbytes + 255) / 256) * 256;
    return p;
  };
  auto allocF  = [&](size_t n) -> float*          { return (float*)allocB(n * 4); };
  auto allocH  = [&](size_t n) -> unsigned short* { return (unsigned short*)allocB(n * 2); };

  // f32 activations
  float* embedded = allocF((size_t)B_ * T_ * H_);
  float* buf1     = allocF((size_t)B_ * T_ * H_);
  float* buf2     = allocF((size_t)B_ * T_ * H_);
  float* gi       = allocF((size_t)B_ * T_ * 3 * H_);
  float* gh       = allocF((size_t)B_ * T_ * 3 * H_);
  float* h0       = allocF((size_t)MDEC * H_);
  float* hdec     = allocF((size_t)MDEC * H_);
  int*   idx      = (int*)allocB(512 * 4);
  // bf16 activations
  unsigned short* embedded_b = allocH((size_t)B_ * T_ * H_);
  unsigned short* buf1_b     = allocH((size_t)B_ * T_ * H_);
  unsigned short* buf2_b     = allocH((size_t)B_ * T_ * H_);
  unsigned short* Gt_b       = allocH((size_t)B_ * T_ * T_);
  unsigned short* curT_b     = allocH((size_t)B_ * H_ * T_);
  unsigned short* wgt_b      = allocH((size_t)B_ * T_ * H_);
  unsigned short* h0_b       = allocH((size_t)MDEC * H_);
  unsigned short* nin_b      = allocH((size_t)MDEC * H_);
  unsigned short* hdec_b     = allocH((size_t)MDEC * H_);
  // bf16 weights
  unsigned short* eWih_b = allocH((size_t)2 * 3 * H_ * H_);
  unsigned short* eWhh_b = allocH((size_t)2 * 3 * H_ * H_);
  unsigned short* dWih_b = allocH((size_t)3 * H_ * H_);
  unsigned short* dWhh_b = allocH((size_t)3 * H_ * H_);
  unsigned short* ffW_b  = allocH((size_t)V_ * H_);

  // One-time weight conversions (ffW_b stays L2-resident: 33 MB < 192 MB L2).
  { int n = 2 * 3 * H_ * H_;
    cvt_bf16<<<(n + 255) / 256, 256, 0, stream>>>(eWih, eWih_b, n);
    cvt_bf16<<<(n + 255) / 256, 256, 0, stream>>>(eWhh, eWhh_b, n); }
  { int n = 3 * H_ * H_;
    cvt_bf16<<<(n + 255) / 256, 256, 0, stream>>>(dWih, dWih_b, n);
    cvt_bf16<<<(n + 255) / 256, 256, 0, stream>>>(dWhh, dWhh_b, n); }
  { int n = V_ * H_;
    cvt_bf16<<<(n + 255) / 256, 256, 0, stream>>>(ffW, ffW_b, n); }

  embed_gather<<<B_ * T_, 256, 0, stream>>>(emb, x, embedded, embedded_b);

  const float*          cur   = embedded;
  const unsigned short* cur_b = embedded_b;
  float*          nxt [2] = {buf1,   buf2};
  unsigned short* nxtb[2] = {buf1_b, buf2_b};
  for (int l = 0; l < 2; ++l) {
    transpose_g  <<<(B_ * T_ * T_) / 256, 256, 0, stream>>>(G, Gt_b, l);
    transpose_act<<<(B_ * H_ * T_) / 256, 256, 0, stream>>>(cur, curT_b);
    // wgt[b] = Gt[b] (128x128) @ cur[b]^T-view (NT, batched) -> bf16 only
    gemm_tdm_bf16<<<dim3(H_ / BN, T_ / BM, B_), 256, 0, stream>>>(
        Gt_b, (size_t)T_ * T_, T_,
        curT_b, (size_t)H_ * T_, T_,
        nullptr, (size_t)T_ * H_, H_, wgt_b, nullptr, T_, H_, T_);
    // gi = wgt @ Wih^T
    gemm_tdm_bf16<<<dim3(3 * H_ / BN, (B_ * T_) / BM, 1), 256, 0, stream>>>(
        wgt_b, 0, H_, eWih_b + (size_t)l * 3 * H_ * H_, 0, H_,
        gi, 0, 3 * H_, nullptr, nullptr, B_ * T_, 3 * H_, H_);
    // gh = cur @ Whh^T
    gemm_tdm_bf16<<<dim3(3 * H_ / BN, (B_ * T_) / BM, 1), 256, 0, stream>>>(
        cur_b, 0, H_, eWhh_b + (size_t)l * 3 * H_ * H_, 0, H_,
        gh, 0, 3 * H_, nullptr, nullptr, B_ * T_, 3 * H_, H_);
    gru_gate<<<(B_ * T_ * H_) / 256, 256, 0, stream>>>(
        gi, gh, ebih + l * 3 * H_, ebhh + l * 3 * H_, cur, nxt[l], nxtb[l], B_ * T_);
    cur = nxt[l]; cur_b = nxtb[l];
  }

  copy_h0     <<<(MDEC * H_ + 255) / 256, 256, 0, stream>>>(cur, h0, h0_b);
  init_next_in<<<(MDEC * H_ + 255) / 256, 256, 0, stream>>>(embedded, nin_b);

  // Decoder gh = h0 @ dec_Whh^T is loop-invariant (h0 never updates).
  gemm_tdm_bf16<<<dim3(3 * H_ / BN, (MDEC + BM - 1) / BM, 1), 256, 0, stream>>>(
      h0_b, 0, H_, dWhh_b, 0, H_, gh, 0, 3 * H_, nullptr, nullptr, MDEC, 3 * H_, H_);

  for (int d = 0; d < D_; ++d) {
    gemm_tdm_bf16<<<dim3(3 * H_ / BN, (MDEC + BM - 1) / BM, 1), 256, 0, stream>>>(
        nin_b, 0, H_, dWih_b, 0, H_, gi, 0, 3 * H_, nullptr, nullptr,
        MDEC, 3 * H_, H_);
    gru_gate<<<(MDEC * H_ + 255) / 256, 256, 0, stream>>>(
        gi, gh, dbih, dbhh, h0, hdec, hdec_b, MDEC);
    // logits -> d_out interleaved: out[m*D*V + d*V + v]
    gemm_tdm_bf16<<<dim3(V_ / BN, (MDEC + BM - 1) / BM, 1), 256, 0, stream>>>(
        hdec_b, 0, H_, ffW_b, 0, H_,
        out + (size_t)d * V_, 0, D_ * V_, nullptr, ffb, MDEC, V_, H_);
    if (d < D_ - 1) {
      argmax_rows<<<MDEC, 256, 0, stream>>>(out + (size_t)d * V_, D_ * V_, idx);
      gather_next<<<MDEC, 256, 0, stream>>>(emb, idx, nin_b);
    }
  }
}
